// DataEmbedding_FeaturePatching_11914239279364
// MI455X (gfx1250) — matrix operand order repaired and checked by reading, not compiled, not run on hardware
//
#include <hip/hip_runtime.h>
#include <hip/hip_bf16.h>

// ---------------------------------------------------------------------------
// DataEmbedding_FeaturePatching for MI455X (gfx1250, wave32, WMMA)
//
// Pipeline:
//   embed_patch_kernel : conv(5/9/15) + mag-pool + patchify -> z f16 (64512,160)
//   cvt_f16_kernel     : fc1_w / fc2_w fp32 -> f16
//   pe_kernel          : 2D sinusoidal positional encoding table (64,63,512) f32
//   gemm_wmma<0>       : h = GELU(z @ fc1_w^T + b1)   -> f16 (64512,2048)
//   gemm_wmma<1>       : out = h @ fc2_w^T + b2 + PE  -> f32 (64512,512)
//
// GEMM: 256 threads = 8 waves (4M x 2N), block tile 256x128, wave tile 64x64
// -> 16 v_wmma per 16 global_load_b128 per K-step (32 FLOP/byte from cache).
// GELU: branchless A&S-7.1.26 erf (|err|<1.5e-7), no exec divergence.
// ---------------------------------------------------------------------------

#define B_ 16
#define L_ 512
#define NCH 64      // 60 + 4
#define NP 63       // patches
#define PATCH_ 16
#define STRIDE_ 8
#define INNER_ 160
#define M_ (B_ * NCH * NP)   // 64512
#define N1_ 2048
#define N2_ 512

typedef __attribute__((ext_vector_type(16))) _Float16 v16h;
typedef __attribute__((ext_vector_type(8)))  float    v8f;

union Frag16 {
    v16h v;
    float4 f4[2];
};

// ---------------------------------------------------------------------------
// Branchless exact-GELU: 0.5*x*(1+erf(x/sqrt2)) with Abramowitz-Stegun 7.1.26
// erf approximation (max abs error 1.5e-7; result truncated to f16 anyway).
// ~12 straight-line VALU ops: v_rcp_f32, v_exp_f32, 6 fma, copysign.
__device__ __forceinline__ float gelu_exact(float x) {
    const float z  = x * 0.70710678118654752f;
    const float az = fabsf(z);
    const float t  = __builtin_amdgcn_rcpf(1.0f + 0.3275911f * az);
    float poly = t * (0.254829592f +
                 t * (-0.284496736f +
                 t * (1.421413741f +
                 t * (-1.453152027f +
                 t * 1.061405429f))));
    const float e = __expf(-az * az);
    const float erf_az = 1.0f - poly * e;
    const float erfz = copysignf(erf_az, z);
    return 0.5f * x * (1.0f + erfz);
}

// ---------------------------------------------------------------------------
// A-fragment (16x32, MxK): lanes 0-15 hold K {0..7, 16..23}, lanes 16-31 hold
// K {8..15, 24..31}; K-pairs contiguous per VGPR -> two contiguous 16B chunks.
__device__ __forceinline__ v16h load_a_frag(const _Float16* __restrict__ A,
                                            int lda, int row, int k0, int lane) {
    Frag16 fr;
    const _Float16* p = A + (long)row * lda + k0 + ((lane >> 4) << 3);
    fr.f4[0] = *reinterpret_cast<const float4*>(p);
    fr.f4[1] = *reinterpret_cast<const float4*>(p + 16);
    return fr.v;
}

// B-fragment (32x16, KxN) with B[k][n] = W[n][k] (W row-major (N,K)):
// lanes 0-15 hold K {0..7, 16..23}, lanes 16-31 hold K {8..15, 24..31}.
__device__ __forceinline__ v16h load_b_frag(const _Float16* __restrict__ W,
                                            int ldw, int col, int k0, int lane) {
    Frag16 fr;
    const _Float16* p = W + (long)col * ldw + k0 + ((lane >> 4) << 3);
    fr.f4[0] = *reinterpret_cast<const float4*>(p);
    fr.f4[1] = *reinterpret_cast<const float4*>(p + 16);
    return fr.v;
}

// ---------------------------------------------------------------------------
// GEMM: C(M x Nn) = A(M x K) * W(Nn x K)^T, f16 in, f32 accumulate.
// EPI 0: bias + exact GELU -> f16 outH.  EPI 1: bias + PE -> f32 outF.
template <int EPI>
__global__ __launch_bounds__(256) void gemm_wmma(
    const _Float16* __restrict__ A, const _Float16* __restrict__ W,
    const float* __restrict__ bias, const float* __restrict__ pe,
    _Float16* __restrict__ outH, float* __restrict__ outF,
    int K, int Nn)
{
    const int lane = threadIdx.x & 31;
    const int wv   = threadIdx.x >> 5;
    const int wm   = wv >> 1;        // 0..3
    const int wn   = wv & 1;         // 0..1
    const int mBase = blockIdx.y * 256 + wm * 64;
    const int nBase = blockIdx.x * 128 + wn * 64;
    const int lr = lane & 15;

    v8f acc[4][4] = {};

    for (int k0 = 0; k0 < K; k0 += 32) {
        v16h a[4], b[4];
#pragma unroll
        for (int i = 0; i < 4; i++) {
            a[i] = load_a_frag(A, K, mBase + i * 16 + lr, k0, lane);
            b[i] = load_b_frag(W, K, nBase + i * 16 + lr, k0, lane);
        }
#pragma unroll
        for (int ai = 0; ai < 4; ai++)
#pragma unroll
            for (int bi = 0; bi < 4; bi++)
                acc[ai][bi] = __builtin_amdgcn_wmma_f32_16x16x32_f16(
                    false, a[ai], false, b[bi], (short)0, acc[ai][bi],
                    false, false);
    }

    // C/D layout: VGPR r -> row r (lanes 0-15) / row r+8 (lanes 16-31); col = lane%16.
    const int rOff = (lane >> 4) << 3;
#pragma unroll
    for (int ai = 0; ai < 4; ai++) {
#pragma unroll
        for (int bi = 0; bi < 4; bi++) {
            const int col = nBase + bi * 16 + lr;
            const float bv = bias[col];
#pragma unroll
            for (int r = 0; r < 8; r++) {
                const long row = (long)mBase + ai * 16 + rOff + r;
                float v = acc[ai][bi][r] + bv;
                if (EPI == 0) {
                    v = gelu_exact(v);
                    outH[row * Nn + col] = (_Float16)v;
                } else {
                    const int m = (int)row;
                    const int p = m % NP;
                    const int n = (m / NP) & (NCH - 1);
                    v += pe[((long)(n * NP + p)) * N2_ + col];
                    outF[row * Nn + col] = v;
                }
            }
        }
    }
}

// ---------------------------------------------------------------------------
// fp32 -> f16 conversion
__global__ void cvt_f16_kernel(const float* __restrict__ in,
                               _Float16* __restrict__ out, int n) {
    int i = blockIdx.x * blockDim.x + threadIdx.x;
    if (i < n) out[i] = (_Float16)in[i];
}

// ---------------------------------------------------------------------------
// PositionalEncoding2D table: pe[n][p][c]; ch=512, channels=256.
// c < 256:  j=c>>1, pos=n;  c >= 256: j=(c-256)>>1, pos=p.
// value = (c odd) ? cos(pos*f_j) : sin(pos*f_j), f_j = 10000^{-(2j)/256}
__global__ void pe_kernel(float* __restrict__ pe) {
    int i = blockIdx.x * blockDim.x + threadIdx.x;   // < 64*63*512
    int c = i & (N2_ - 1);
    int rem = i >> 9;            // n*63 + p
    int p = rem % NP;
    int n = rem / NP;
    int cc = (c < 256) ? c : (c - 256);
    int j = cc >> 1;
    float pos = (c < 256) ? (float)n : (float)p;
    float freq = expf(-9.210340371976184f * ((float)(2 * j) / 256.0f));
    float s = pos * freq;
    pe[i] = (c & 1) ? cosf(s) : sinf(s);
}

// ---------------------------------------------------------------------------
// Front-end: per (b,n) series -> conv branches + mag-pool + patchify -> z f16.
// One block per (b,n); series cached in LDS; conv recomputed per pool window
// (<=105 fmas per output element, negligible vs the GEMMs).
__global__ __launch_bounds__(256) void embed_patch_kernel(
    const float* __restrict__ x,  const float* __restrict__ xm,
    const float* __restrict__ w1, const float* __restrict__ b1,
    const float* __restrict__ w2, const float* __restrict__ b2,
    const float* __restrict__ w3, const float* __restrict__ b3,
    _Float16* __restrict__ z)
{
    __shared__ float s[L_];
    const int bn = blockIdx.x;
    const int b = bn >> 6, n = bn & 63;
    const int tid = threadIdx.x;

    for (int l = tid; l < L_; l += 256)
        s[l] = (n < 60) ? x[((long)b * L_ + l) * 60 + n]
                        : xm[((long)b * L_ + l) * 4 + (n - 60)];
    __syncthreads();

    const long rowBase = (long)bn * NP;

    // channel 0: raw series patches (feature index t)
    for (int i = tid; i < NP * PATCH_; i += 256) {
        int p = i >> 4, t = i & 15;
        z[(rowBase + p) * INNER_ + t] = (_Float16)s[p * STRIDE_ + t];
    }

    const float* Ws[3] = {w1, w2, w3};
    const float* Bs[3] = {b1, b2, b3};
    const int   Kc[3] = {5, 9, 15};
    const int   PZ[3] = {2, 4, 6};   // left zero pad; pool window = PZ+1

    for (int br = 0; br < 3; br++) {
        const int K = Kc[br], pz = PZ[br];
        const float* w = Ws[br];
        const float* bb = Bs[br];
        // per-branch z elements: 63 patches x 16 pos x 3 ch
        for (int i = tid; i < NP * PATCH_ * 3; i += 256) {
            int c = i / (NP * PATCH_);
            int pi = i % (NP * PATCH_);
            int p = pi >> 4, t = pi & 15;
            int tt = p * STRIDE_ + t;           // pooled position in 0..511
            float best = 0.0f, bestAbs = -1.0f;
            // magnitude max over ypad[tt .. tt+pz] -> y indices tt-pz .. tt
            for (int j = tt - pz; j <= tt; j++) {
                float v;
                if (j < 0) {
                    v = 0.0f;                   // zero pad
                } else {
                    float a = bb[c];
                    for (int q = 0; q < K; q++) {
                        int idx = j + q - (K - 1);  // edge pad left by K-1
                        a += w[c * K + q] * s[idx < 0 ? 0 : idx];
                    }
                    v = a;
                }
                float av = fabsf(v);
                if (av > bestAbs) { bestAbs = av; best = v; }  // first-max tie
            }
            z[(rowBase + p) * INNER_ + (1 + br * 3 + c) * PATCH_ + t] = (_Float16)best;
        }
    }
}

// ---------------------------------------------------------------------------
extern "C" void kernel_launch(void* const* d_in, const int* in_sizes, int n_in,
                              void* d_out, int out_size, void* d_ws, size_t ws_size,
                              hipStream_t stream) {
    const float* x     = (const float*)d_in[0];
    const float* xmark = (const float*)d_in[1];
    const float* w1    = (const float*)d_in[2];
    const float* b1    = (const float*)d_in[3];
    const float* w2    = (const float*)d_in[4];
    const float* b2    = (const float*)d_in[5];
    const float* w3    = (const float*)d_in[6];
    const float* b3    = (const float*)d_in[7];
    const float* fc1_w = (const float*)d_in[8];
    const float* fc1_b = (const float*)d_in[9];
    const float* fc2_w = (const float*)d_in[10];
    const float* fc2_b = (const float*)d_in[11];
    float* out = (float*)d_out;

    // workspace layout (all offsets 16B aligned)
    char* ws = (char*)d_ws;
    size_t off = 0;
    _Float16* zf  = (_Float16*)(ws + off); off += (size_t)M_ * INNER_ * 2;        // 20,643,840
    _Float16* w1f = (_Float16*)(ws + off); off += (size_t)N1_ * INNER_ * 2;       //    655,360
    _Float16* w2f = (_Float16*)(ws + off); off += (size_t)N2_ * N1_ * 2;          //  2,097,152
    float*    pe  = (float*)   (ws + off); off += (size_t)NCH * NP * N2_ * 4;     //  8,257,536
    _Float16* h   = (_Float16*)(ws + off);                                        // 264,241,152

    // prep
    cvt_f16_kernel<<<(N1_ * INNER_ + 255) / 256, 256, 0, stream>>>(fc1_w, w1f, N1_ * INNER_);
    cvt_f16_kernel<<<(N2_ * N1_ + 255) / 256, 256, 0, stream>>>(fc2_w, w2f, N2_ * N1_);
    pe_kernel<<<(NCH * NP * N2_) / 256, 256, 0, stream>>>(pe);
    embed_patch_kernel<<<B_ * NCH, 256, 0, stream>>>(x, xmark, w1, b1, w2, b2, w3, b3, zf);

    // fc1: (64512 x 160) x (160 x 2048), GELU, f16 out
    gemm_wmma<0><<<dim3(N1_ / 128, M_ / 256), 256, 0, stream>>>(
        zf, w1f, fc1_b, nullptr, h, nullptr, INNER_, N1_);
    // fc2: (64512 x 2048) x (2048 x 512), bias + PE, f32 out
    gemm_wmma<1><<<dim3(N2_ / 128, M_ / 256), 256, 0, stream>>>(
        h, w2f, fc2_b, pe, nullptr, out, N1_, N2_);
}
